// DAGMLP_46033459478957
// MI455X (gfx1250) — matrix operand order repaired and checked by reading, not compiled, not run on hardware
//
#include <hip/hip_runtime.h>

// Problem constants (match the reference harness)
#define N_NODES 100000
#define N_EDGES 600000
#define F_IN    64
#define D_HID   128
#define N_LAY   3
#define N_GRAPH 256
#define T_OUT   10
#define N_LEAVES 50000
#define N_READ   20000
#define BN_EPS  1e-5f

typedef __attribute__((ext_vector_type(16))) _Float16 v16h;
typedef __attribute__((ext_vector_type(8)))  _Float16 v8h;
typedef __attribute__((ext_vector_type(8)))  float    v8f;

// ---------------------------------------------------------------------------
// Generic zero-fill
// ---------------------------------------------------------------------------
__global__ void dagmlp_zero_kernel(float* __restrict__ p, long long n) {
    long long i  = (long long)blockIdx.x * blockDim.x + threadIdx.x;
    long long st = (long long)gridDim.x * blockDim.x;
    for (; i < n; i += st) p[i] = 0.0f;
}

// ---------------------------------------------------------------------------
// WMMA GEMM:  C[M x 128] = A[M x K] @ W[K x 128] + bias
// Block: 256 threads (8 waves). Block tile 128 rows x 128 cols.
// Wave w computes rows [16w, 16w+16) x 128 cols = 8 WMMA 16x16 tiles.
// K must be a multiple of 32 (K = 64 or 128 here).
// A, W are f32 in global memory; converted to f16 when staged to LDS.
// LDS rows padded to 40 halves (80 B): 16B-aligned and bank-conflict-free.
// ---------------------------------------------------------------------------
#define TILE_M 128
#define TILE_K 32
#define LDS_PITCH 40

__global__ __launch_bounds__(256)
void dagmlp_gemm_bias_kernel(const float* __restrict__ A,
                             const float* __restrict__ W,
                             const float* __restrict__ bias,
                             float* __restrict__ C,
                             int M, int K) {
    __shared__ __align__(16) _Float16 a_sh[TILE_M][LDS_PITCH]; // row-major
    __shared__ __align__(16) _Float16 w_sh[D_HID][LDS_PITCH];  // transposed [col][k]

    const int tid  = threadIdx.x;
    const int wave = tid >> 5;
    const int lane = tid & 31;
    const int row0 = blockIdx.x * TILE_M;

    v8f acc[8];
#pragma unroll
    for (int t = 0; t < 8; ++t)
#pragma unroll
        for (int e = 0; e < 8; ++e) acc[t][e] = 0.0f;

    // staging coordinates (constant across K loop)
    const int ar   = tid >> 1;             // A row 0..127
    const int aks  = (tid & 1) << 4;       // A k-segment: 0 or 16
    const int wkk  = tid >> 3;             // W k row 0..31
    const int wcs  = (tid & 7) << 4;       // W col segment: 0,16,...,112

    for (int k0 = 0; k0 < K; k0 += TILE_K) {
        // ---- stage A tile (128 x 32): 16 contiguous elems per thread
        {
            int gr = row0 + ar;
            v8h lo, hi;
            if (gr < M) {
                const float* ap = &A[(long long)gr * K + k0 + aks];
                float4 v0 = *(const float4*)(ap + 0);
                float4 v1 = *(const float4*)(ap + 4);
                float4 v2 = *(const float4*)(ap + 8);
                float4 v3 = *(const float4*)(ap + 12);
                lo[0] = (_Float16)v0.x; lo[1] = (_Float16)v0.y;
                lo[2] = (_Float16)v0.z; lo[3] = (_Float16)v0.w;
                lo[4] = (_Float16)v1.x; lo[5] = (_Float16)v1.y;
                lo[6] = (_Float16)v1.z; lo[7] = (_Float16)v1.w;
                hi[0] = (_Float16)v2.x; hi[1] = (_Float16)v2.y;
                hi[2] = (_Float16)v2.z; hi[3] = (_Float16)v2.w;
                hi[4] = (_Float16)v3.x; hi[5] = (_Float16)v3.y;
                hi[6] = (_Float16)v3.z; hi[7] = (_Float16)v3.w;
            } else {
#pragma unroll
                for (int e = 0; e < 8; ++e) { lo[e] = (_Float16)0.0f; hi[e] = (_Float16)0.0f; }
            }
            v8h* dp = (v8h*)&a_sh[ar][aks];
            dp[0] = lo;
            dp[1] = hi;
        }
        // ---- stage W tile transposed (32 x 128 -> [col][k])
        {
            const float* wp = &W[(long long)(k0 + wkk) * D_HID + wcs];
#pragma unroll
            for (int e = 0; e < 16; e += 4) {
                float4 v = *(const float4*)(wp + e);
                w_sh[wcs + e + 0][wkk] = (_Float16)v.x;
                w_sh[wcs + e + 1][wkk] = (_Float16)v.y;
                w_sh[wcs + e + 2][wkk] = (_Float16)v.z;
                w_sh[wcs + e + 3][wkk] = (_Float16)v.w;
            }
        }
        // ---- prefetch next A K-tile (global_prefetch_b8 on gfx1250)
        if (k0 + TILE_K < K) {
            int pr = row0 + ar;
            if (pr < M)
                __builtin_prefetch(&A[(long long)pr * K + k0 + TILE_K + aks], 0, 1);
        }
        __syncthreads();

        // ---- A fragment, 16x32 f16 layout:
        // lanes 0-15: row M = lane, K = 0..15
        // lanes 16-31: row M = lane-16, K = 16..31
        v16h afrag;
        {
            int m  = lane & 15;
            int kb = (lane < 16) ? 0 : 16;
            const _Float16* ap = &a_sh[wave * 16 + m][0];
#pragma unroll
            for (int e = 0; e < 8; ++e) afrag[e] = ap[kb + e];
#pragma unroll
            for (int e = 0; e < 8; ++e) afrag[8 + e] = ap[kb + 8 + e];
        }

        // ---- 8 column tiles; B fragment 32x16 f16 layout:
        // col N = lane&15; lanes 0-15 hold K=0..15, lanes 16-31 hold K=16..31
        int n  = lane & 15;
        int kb = (lane < 16) ? 0 : 16;
#pragma unroll
        for (int t = 0; t < 8; ++t) {
            v16h bfrag;
            const _Float16* bp = &w_sh[t * 16 + n][0];
#pragma unroll
            for (int e = 0; e < 16; ++e) bfrag[e] = bp[kb + e];
            acc[t] = __builtin_amdgcn_wmma_f32_16x16x32_f16(
                false, afrag, false, bfrag,
                (short)0, acc[t], false, false);
        }
        __syncthreads();
    }

    // ---- store: C/D layout: col = lane&15, row = VGPR r + 8*(lane>=16)
    int n     = lane & 15;
    int rbase = row0 + wave * 16 + ((lane >> 4) << 3);
#pragma unroll
    for (int t = 0; t < 8; ++t) {
        int col = t * 16 + n;
        float bb = bias ? bias[col] : 0.0f;
#pragma unroll
        for (int r = 0; r < 8; ++r) {
            int gr = rbase + r;
            if (gr < M) C[(long long)gr * D_HID + col] = acc[t][r] + bb;
        }
    }
}

// ---------------------------------------------------------------------------
// BatchNorm stats: stats[0..127]=sum, [128..255]=sumsq, [256]=count.
// tmask == nullptr -> all rows; else rows with tmask[r] > 0 (binarized).
// ---------------------------------------------------------------------------
__global__ __launch_bounds__(256)
void dagmlp_bn_stats_kernel(const float* __restrict__ x,
                            const float* __restrict__ tmask,
                            float* __restrict__ stats, int M) {
    int c    = threadIdx.x & 127;
    int half = threadIdx.x >> 7;
    float s = 0.0f, ss = 0.0f, cntp = 0.0f;
    for (int r = blockIdx.x * 2 + half; r < M; r += gridDim.x * 2) {
        float m = tmask ? ((tmask[r] > 0.0f) ? 1.0f : 0.0f) : 1.0f;
        float v = x[(long long)r * D_HID + c];
        s  += m * v;
        ss += m * v * v;
        if (c == 0) cntp += m;
    }
    atomicAdd(&stats[c], s);
    atomicAdd(&stats[D_HID + c], ss);
    if (c == 0) atomicAdd(&stats[2 * D_HID], cntp);
}

// In-place normalize + ReLU over all rows. Grid stride is a multiple of 128,
// so the column (and its BN constants) is loop-invariant per thread.
__global__ void dagmlp_bn_apply_relu_kernel(float* __restrict__ x,
                                            const float* __restrict__ stats,
                                            const float* __restrict__ g,
                                            const float* __restrict__ b,
                                            long long n) {
    long long i0 = (long long)blockIdx.x * blockDim.x + threadIdx.x;
    long long st = (long long)gridDim.x * blockDim.x;
    int c = (int)(i0 & 127);
    float cnt  = fmaxf(stats[2 * D_HID], 1.0f);
    float mean = stats[c] / cnt;
    float var  = stats[D_HID + c] / cnt - mean * mean;
    float sc   = g[c] * rsqrtf(var + BN_EPS);
    float sh   = b[c] - mean * sc;
    for (long long i = i0; i < n; i += st)
        x[i] = fmaxf(x[i] * sc + sh, 0.0f);
}

// ---------------------------------------------------------------------------
// x[leaves] = feature[leaves] (x pre-zeroed)
// ---------------------------------------------------------------------------
__global__ void dagmlp_leaves_kernel(const int* __restrict__ leaves,
                                     const float* __restrict__ feat,
                                     float* __restrict__ x, long long n) {
    long long i0 = (long long)blockIdx.x * blockDim.x + threadIdx.x;
    long long st = (long long)gridDim.x * blockDim.x;
    int c = (int)(i0 & 127);
    for (long long i = i0; i < n; i += st) {
        long long node = leaves[i >> 7];
        x[node * D_HID + c] = feat[node * D_HID + c];
    }
}

// ---------------------------------------------------------------------------
// Edge gather-scatter: EX[dst] += w * x[src] for active edges; tcnt[dst]++.
// 2 edges per 256-thread block (128 dims each).
// ---------------------------------------------------------------------------
__global__ __launch_bounds__(256)
void dagmlp_edge_scatter_kernel(const float* __restrict__ x,
                                const int* __restrict__ src,
                                const int* __restrict__ dst,
                                const float* __restrict__ mult,
                                const int* __restrict__ lmask,
                                int layer,
                                float* __restrict__ EX,
                                float* __restrict__ tcnt, int nedges) {
    int e = blockIdx.x * 2 + (threadIdx.x >> 7);
    int c = threadIdx.x & 127;
    if (e >= nedges) return;
    if (lmask[e] != layer) return;
    long long sN = src[e];
    long long dN = dst[e];
    float w = mult[e];
    atomicAdd(&EX[dN * D_HID + c], x[sN * D_HID + c] * w);
    if (c == 0) atomicAdd(&tcnt[dN], 1.0f);
}

// s = is_t * (1+eps) * feature + EX   (s holds EX, add in place)
__global__ void dagmlp_make_s_kernel(const float* __restrict__ feat,
                                     const float* __restrict__ tcnt,
                                     const float* __restrict__ dag_eps,
                                     int layer, float* __restrict__ s,
                                     long long n) {
    long long i  = (long long)blockIdx.x * blockDim.x + threadIdx.x;
    long long st = (long long)gridDim.x * blockDim.x;
    float scale = 1.0f + dag_eps[layer];
    for (; i < n; i += st) {
        int r = (int)(i >> 7);
        if (tcnt[r] > 0.0f) s[i] += scale * feat[i];
    }
}

// x += where(is_t, h, s)
__global__ void dagmlp_update_x_kernel(float* __restrict__ x,
                                       const float* __restrict__ h,
                                       const float* __restrict__ s,
                                       const float* __restrict__ tcnt,
                                       long long n) {
    long long i  = (long long)blockIdx.x * blockDim.x + threadIdx.x;
    long long st = (long long)gridDim.x * blockDim.x;
    for (; i < n; i += st) {
        int r = (int)(i >> 7);
        x[i] += (tcnt[r] > 0.0f) ? h[i] : s[i];
    }
}

// ---------------------------------------------------------------------------
// Readout: pool[batch[idx]] += x[idx] over all (L+1)*N_READ indices.
// ---------------------------------------------------------------------------
__global__ void dagmlp_readout_kernel(const int* __restrict__ reads,
                                      const int* __restrict__ batch,
                                      const float* __restrict__ x,
                                      float* __restrict__ pool, long long n) {
    long long i0 = (long long)blockIdx.x * blockDim.x + threadIdx.x;
    long long st = (long long)gridDim.x * blockDim.x;
    int c = (int)(i0 & 127);
    for (long long i = i0; i < n; i += st) {
        long long node = reads[i >> 7];
        int g = batch[node];
        atomicAdd(&pool[(long long)g * D_HID + c], x[node * D_HID + c]);
    }
}

// out[g,t] = 0.25 * sum_c pool[g,c] * Wl[c,t] + bl[t]
__global__ __launch_bounds__(256)
void dagmlp_final_kernel(const float* __restrict__ pool,
                         const float* __restrict__ Wl,
                         const float* __restrict__ bl,
                         float* __restrict__ out) {
    int i = blockIdx.x * blockDim.x + threadIdx.x;
    if (i >= N_GRAPH * T_OUT) return;
    int g = i / T_OUT;
    int t = i % T_OUT;
    float acc = 0.0f;
#pragma unroll 4
    for (int c = 0; c < D_HID; ++c)
        acc += pool[(long long)g * D_HID + c] * Wl[c * T_OUT + t];
    out[i] = acc * 0.25f + bl[t];
}

// ---------------------------------------------------------------------------
// Host orchestration
// ---------------------------------------------------------------------------
static inline int dagmlp_gs_blocks(long long n) {
    long long b = (n + 255) / 256;
    if (b > 65535) b = 65535;
    if (b < 1) b = 1;
    return (int)b;
}

extern "C" void kernel_launch(void* const* d_in, const int* in_sizes, int n_in,
                              void* d_out, int out_size, void* d_ws, size_t ws_size,
                              hipStream_t stream) {
    (void)in_sizes; (void)n_in; (void)out_size; (void)ws_size;

    const float* dag_x = (const float*)d_in[0];
    const int*   eidx  = (const int*)d_in[1];    // [2,E]: src then dst
    const float* mult  = (const float*)d_in[2];
    const int*   lmask = (const int*)d_in[3];
    const int*   batch = (const int*)d_in[4];
    const int*   leaves= (const int*)d_in[5];
    const int*   reads = (const int*)d_in[6];    // [(L+1)*N_READ]
    const float* W1    = (const float*)d_in[7];
    const float* b1    = (const float*)d_in[8];
    const float* g1    = (const float*)d_in[9];
    const float* be1   = (const float*)d_in[10];
    const float* W2    = (const float*)d_in[11];
    const float* b2    = (const float*)d_in[12];
    const float* g2    = (const float*)d_in[13];
    const float* be2   = (const float*)d_in[14];
    const float* dW1   = (const float*)d_in[15];
    const float* db1   = (const float*)d_in[16];
    const float* dg1   = (const float*)d_in[17];
    const float* dbe1  = (const float*)d_in[18];
    const float* dW2   = (const float*)d_in[19];
    const float* db2   = (const float*)d_in[20];
    const float* dg2   = (const float*)d_in[21];
    const float* dbe2  = (const float*)d_in[22];
    const float* deps  = (const float*)d_in[23];
    const float* Wl    = (const float*)d_in[24];
    const float* bl    = (const float*)d_in[25];
    float* out = (float*)d_out;

    const int* esrc = eidx;
    const int* edst = eidx + N_EDGES;

    const long long ND = (long long)N_NODES * D_HID;
    float* ws      = (float*)d_ws;
    float* feature = ws;            // [N,D]
    float* x       = ws + 1 * ND;   // [N,D]
    float* s       = ws + 2 * ND;   // [N,D]  (also E_X accumulator)
    float* t1      = ws + 3 * ND;   // [N,D]
    float* t2      = ws + 4 * ND;   // [N,D]
    float* tcnt    = ws + 5 * ND;   // [N]
    float* stats   = tcnt + N_NODES;            // [272]
    float* pool    = stats + 272;               // [G,D]

    const int gemmBlocks = (N_NODES + TILE_M - 1) / TILE_M;
    const int statBlocks = 1024;

    // ---- node feature transformation: relu(BN(x@W1+b1)) -> relu(BN(.@W2+b2))
    dagmlp_gemm_bias_kernel<<<gemmBlocks, 256, 0, stream>>>(dag_x, W1, b1, t1,
                                                           N_NODES, F_IN);
    dagmlp_zero_kernel<<<1, 256, 0, stream>>>(stats, 272);
    dagmlp_bn_stats_kernel<<<statBlocks, 256, 0, stream>>>(t1, nullptr, stats, N_NODES);
    dagmlp_bn_apply_relu_kernel<<<dagmlp_gs_blocks(ND), 256, 0, stream>>>(t1, stats, g1, be1, ND);

    dagmlp_gemm_bias_kernel<<<gemmBlocks, 256, 0, stream>>>(t1, W2, b2, feature,
                                                           N_NODES, D_HID);
    dagmlp_zero_kernel<<<1, 256, 0, stream>>>(stats, 272);
    dagmlp_bn_stats_kernel<<<statBlocks, 256, 0, stream>>>(feature, nullptr, stats, N_NODES);
    dagmlp_bn_apply_relu_kernel<<<dagmlp_gs_blocks(ND), 256, 0, stream>>>(feature, stats, g2, be2, ND);

    // ---- x = zeros; x[leaves0] = feature[leaves0]
    dagmlp_zero_kernel<<<dagmlp_gs_blocks(ND), 256, 0, stream>>>(x, ND);
    const long long nleaf = (long long)N_LEAVES * D_HID;
    dagmlp_leaves_kernel<<<dagmlp_gs_blocks(nleaf), 256, 0, stream>>>(leaves, feature, x, nleaf);

    // ---- DAG layers
    for (int layer = 0; layer < N_LAY; ++layer) {
        dagmlp_zero_kernel<<<dagmlp_gs_blocks(ND), 256, 0, stream>>>(s, ND);
        dagmlp_zero_kernel<<<dagmlp_gs_blocks(N_NODES), 256, 0, stream>>>(tcnt, N_NODES);

        dagmlp_edge_scatter_kernel<<<(N_EDGES + 1) / 2, 256, 0, stream>>>(
            x, esrc, edst, mult, lmask, layer, s, tcnt, N_EDGES);

        dagmlp_make_s_kernel<<<dagmlp_gs_blocks(ND), 256, 0, stream>>>(
            feature, tcnt, deps, layer, s, ND);

        const float* lW1  = dW1  + (long long)layer * D_HID * D_HID;
        const float* lb1  = db1  + (long long)layer * D_HID;
        const float* lg1  = dg1  + (long long)layer * D_HID;
        const float* lbe1 = dbe1 + (long long)layer * D_HID;
        const float* lW2  = dW2  + (long long)layer * D_HID * D_HID;
        const float* lb2  = db2  + (long long)layer * D_HID;
        const float* lg2  = dg2  + (long long)layer * D_HID;
        const float* lbe2 = dbe2 + (long long)layer * D_HID;

        dagmlp_gemm_bias_kernel<<<gemmBlocks, 256, 0, stream>>>(s, lW1, lb1, t1,
                                                               N_NODES, D_HID);
        dagmlp_zero_kernel<<<1, 256, 0, stream>>>(stats, 272);
        dagmlp_bn_stats_kernel<<<statBlocks, 256, 0, stream>>>(t1, tcnt, stats, N_NODES);
        dagmlp_bn_apply_relu_kernel<<<dagmlp_gs_blocks(ND), 256, 0, stream>>>(t1, stats, lg1, lbe1, ND);

        dagmlp_gemm_bias_kernel<<<gemmBlocks, 256, 0, stream>>>(t1, lW2, lb2, t2,
                                                               N_NODES, D_HID);
        dagmlp_zero_kernel<<<1, 256, 0, stream>>>(stats, 272);
        dagmlp_bn_stats_kernel<<<statBlocks, 256, 0, stream>>>(t2, tcnt, stats, N_NODES);
        dagmlp_bn_apply_relu_kernel<<<dagmlp_gs_blocks(ND), 256, 0, stream>>>(t2, stats, lg2, lbe2, ND);

        dagmlp_update_x_kernel<<<dagmlp_gs_blocks(ND), 256, 0, stream>>>(x, t2, s, tcnt, ND);
    }

    // ---- readout: single pooled accumulation over all heights, mean folded as 0.25
    const long long poolN = (long long)N_GRAPH * D_HID;
    dagmlp_zero_kernel<<<dagmlp_gs_blocks(poolN), 256, 0, stream>>>(pool, poolN);
    const long long nread = (long long)(N_LAY + 1) * N_READ * D_HID;
    dagmlp_readout_kernel<<<dagmlp_gs_blocks(nread), 256, 0, stream>>>(reads, batch, x, pool, nread);

    dagmlp_final_kernel<<<(N_GRAPH * T_OUT + 255) / 256, 256, 0, stream>>>(pool, Wl, bl, out);
}
